// TrajectoryPredictor_12807592476996
// MI455X (gfx1250) — matrix operand order repaired and checked by reading, not compiled, not run on hardware
//
#include <hip/hip_runtime.h>
#include <stdint.h>
#include <stddef.h>

#define BB   8192
#define TT   100
#define INF  16
#define HH   128
#define OUTD 2
#define TGTN 60
#define BT   64

typedef __attribute__((ext_vector_type(16))) __bf16 v16bf;
typedef __attribute__((ext_vector_type(8)))  float  v8f;
typedef __attribute__((ext_vector_type(4)))  int    v4i;

struct __align__(16) Q16 { unsigned int x, y, z, w; };
struct Q32 { Q16 lo, hi; };

// ---------------- async global->LDS support (gfx1250) ----------------
#if defined(__has_builtin)
#if __has_builtin(__builtin_amdgcn_global_load_async_to_lds_b128)
#define HAVE_ASYNC_LDS 1
#endif
#if __has_builtin(__builtin_amdgcn_s_wait_asynccnt)
#define HAVE_WAIT_ASYNC 1
#endif
#endif

__device__ __forceinline__ void async_copy16(const unsigned short* g, unsigned short* l) {
#ifdef HAVE_ASYNC_LDS
  __builtin_amdgcn_global_load_async_to_lds_b128(
      (__attribute__((address_space(1))) v4i*)g,
      (__attribute__((address_space(3))) v4i*)l, 0, 0);
#else
  *(Q16*)l = *(const Q16*)g;
#endif
}
__device__ __forceinline__ void wait_async() {
#ifdef HAVE_ASYNC_LDS
#ifdef HAVE_WAIT_ASYNC
  __builtin_amdgcn_s_wait_asynccnt(0);
#else
  asm volatile("s_wait_asynccnt 0" ::: "memory");
#endif
#endif
}

// ---------------- workspace layout (bytes) ----------------
static constexpr size_t WC0_OFF  = 0;                                  // enc L0 concat W  [512][160] bf16
static constexpr size_t WC1_OFF  = WC0_OFF + (size_t)512*160*2;        // enc L1 concat W  [512][256] bf16
static constexpr size_t B0_OFF   = WC1_OFF + (size_t)512*256*2;        // enc L0 bias f32[512]
static constexpr size_t B1_OFF   = B0_OFF + 512*4;                     // enc L1 bias
static constexpr size_t DHH0_OFF = B1_OFF + 512*4;                     // dec L0 Whh [512][128] bf16
static constexpr size_t DC1_OFF  = DHH0_OFF + (size_t)512*128*2;       // dec L1 concat W [512][256] bf16
static constexpr size_t DB0_OFF  = DC1_OFF + (size_t)512*256*2;        // dec L0 bias
static constexpr size_t DB1_OFF  = DB0_OFF + 512*4;                    // dec L1 bias
static constexpr size_t H0F_OFF  = DB1_OFF + 512*4;                    // enc L0 final h f32 [8192][128]
static constexpr size_t C0F_OFF  = H0F_OFF + (size_t)BB*HH*4;
static constexpr size_t H1F_OFF  = C0F_OFF + (size_t)BB*HH*4;
static constexpr size_t C1F_OFF  = H1F_OFF + (size_t)BB*HH*4;
static constexpr size_t XBF_OFF  = C1F_OFF + (size_t)BB*HH*4;          // x in bf16 [8192][100][16]
// end = XBF_OFF + 26,214,400  (~44 MB total)

// ---------------- LDS layouts ----------------
static constexpr int ENC_SW0   = 0;                        // [512][168] ushort
static constexpr int ENC_SCAT  = 512*168*2;                // [64][264] ushort  (cols 0..127 h0, 128..255 h1)
static constexpr int ENC_SXZ   = ENC_SCAT + 64*264*2;      // [2][64][32] ushort (double-buffered x | zero pad)
static constexpr int ENC_SB0   = ENC_SXZ + 2*64*32*2;
static constexpr int ENC_SB1   = ENC_SB0 + 512*4;
static constexpr int ENC_LDS   = ENC_SB1 + 512*4;          // 218112 bytes

static constexpr int DEC_SW1   = 0;                        // [512][264] ushort
static constexpr int DEC_SCAT  = 512*264*2;                // [64][264] ushort
static constexpr int DEC_SOUT  = DEC_SCAT + 64*264*2;      // float [64][2]
static constexpr int DEC_SFC   = DEC_SOUT + 64*2*4;        // float [2][128]
static constexpr int DEC_SFCB  = DEC_SFC + 2*128*4;        // float[2] (+pad)
static constexpr int DEC_SB0   = DEC_SFCB + 16;
static constexpr int DEC_SB1   = DEC_SB0 + 512*4;
static constexpr int DEC_LDS   = DEC_SB1 + 512*4;          // 309776 bytes

// ---------------- helpers ----------------
__device__ __forceinline__ unsigned short f2bf(float f) {
  unsigned int u = __builtin_bit_cast(unsigned int, f);
  u += 0x7FFFu + ((u >> 16) & 1u);
  return (unsigned short)(u >> 16);
}
__device__ __forceinline__ float bf2f(unsigned short h) {
  unsigned int u = ((unsigned int)h) << 16;
  return __builtin_bit_cast(float, u);
}
__device__ __forceinline__ float sigm(float x) { return 1.f / (1.f + __expf(-x)); }
__device__ __forceinline__ float tanh_f(float x) { float e = __expf(2.f * x); return (e - 1.f) / (e + 1.f); }
__device__ __forceinline__ v8f vzero() {
  v8f z;
#pragma unroll
  for (int i = 0; i < 8; ++i) z[i] = 0.f;
  return z;
}
__device__ __forceinline__ v8f wmma_bf16(v16bf a, v16bf b, v8f c) {
  return __builtin_amdgcn_wmma_f32_16x16x32_bf16(false, a, false, b, (short)0, c, false, false);
}
// A-frag (16x32): lane row = row; two 16B chunks at e0, e0+16 (e0 already includes hi*8)
__device__ __forceinline__ v16bf loadAfrag(const unsigned short* base, int stride, int row, int e0) {
  Q32 q;
  q.lo = *(const Q16*)(base + row * stride + e0);
  q.hi = *(const Q16*)(base + row * stride + e0 + 16);
  return __builtin_bit_cast(v16bf, q);
}
// B-frag (32x16): lane n = W row; two 16B chunks at e0, e0+16 (e0 already includes hi*8)
__device__ __forceinline__ v16bf loadBfrag(const unsigned short* base, int stride, int n, int e0) {
  Q32 q;
  q.lo = *(const Q16*)(base + n * stride + e0);
  q.hi = *(const Q16*)(base + n * stride + e0 + 16);
  return __builtin_bit_cast(v16bf, q);
}

// ---------------- weight & input prep ----------------
__global__ __launch_bounds__(256) void prep_kernel(
    const float* x,
    const float* eWih0, const float* eWhh0, const float* ebih0, const float* ebhh0,
    const float* eWih1, const float* eWhh1, const float* ebih1, const float* ebhh1,
    const float* dWhh0, const float* dWih1, const float* dWhh1,
    const float* dbih0, const float* dbhh0, const float* dbih1, const float* dbhh1,
    char* ws) {
  unsigned short* Wc0 = (unsigned short*)(ws + WC0_OFF);
  unsigned short* Wc1 = (unsigned short*)(ws + WC1_OFF);
  float* b0 = (float*)(ws + B0_OFF);
  float* b1 = (float*)(ws + B1_OFF);
  unsigned short* Dh0 = (unsigned short*)(ws + DHH0_OFF);
  unsigned short* Dc1 = (unsigned short*)(ws + DC1_OFF);
  float* db0 = (float*)(ws + DB0_OFF);
  float* db1 = (float*)(ws + DB1_OFF);
  unsigned short* xb = (unsigned short*)(ws + XBF_OFF);
  int tid = blockIdx.x * blockDim.x + threadIdx.x;
  int np = gridDim.x * blockDim.x;
  // x -> bf16
  for (int i = tid; i < BB * TT * INF; i += np) xb[i] = f2bf(x[i]);
  // enc L0 concat: cols 0..127 = W_hh0, 128..143 = W_ih0, 144..159 = 0
  for (int i = tid; i < 512 * 160; i += np) {
    int n = i / 160, k = i % 160;
    float v = (k < 128) ? eWhh0[n * 128 + k] : ((k < 144) ? eWih0[n * 16 + (k - 128)] : 0.f);
    Wc0[i] = f2bf(v);
  }
  // enc L1 concat: cols 0..127 = W_ih1 (h0 side), 128..255 = W_hh1
  for (int i = tid; i < 512 * 256; i += np) {
    int n = i >> 8, k = i & 255;
    float v = (k < 128) ? eWih1[n * 128 + k] : eWhh1[n * 128 + (k - 128)];
    Wc1[i] = f2bf(v);
  }
  for (int i = tid; i < 512 * 128; i += np) Dh0[i] = f2bf(dWhh0[i]);
  // dec L1 concat: cols 0..127 = dW_ih1 (dh0 side), 128..255 = dW_hh1
  for (int i = tid; i < 512 * 256; i += np) {
    int n = i >> 8, k = i & 255;
    float v = (k < 128) ? dWih1[n * 128 + k] : dWhh1[n * 128 + (k - 128)];
    Dc1[i] = f2bf(v);
  }
  for (int i = tid; i < 512; i += np) {
    b0[i] = ebih0[i] + ebhh0[i];
    b1[i] = ebih1[i] + ebhh1[i];
    db0[i] = dbih0[i] + dbhh0[i];
    db1[i] = dbih1[i] + dbhh1[i];
  }
}

// ---------------- fused 2-layer LSTM encoder ----------------
__global__ __launch_bounds__(256) void enc_kernel(char* __restrict__ ws) {
  extern __shared__ char smem[];
  unsigned short* sW0  = (unsigned short*)(smem + ENC_SW0);
  unsigned short* sCat = (unsigned short*)(smem + ENC_SCAT);
  unsigned short* sXZ  = (unsigned short*)(smem + ENC_SXZ);
  float* sB0 = (float*)(smem + ENC_SB0);
  float* sB1 = (float*)(smem + ENC_SB1);

  const unsigned short* gW0 = (const unsigned short*)(ws + WC0_OFF);
  const unsigned short* gW1 = (const unsigned short*)(ws + WC1_OFF);
  const float* gb0 = (const float*)(ws + B0_OFF);
  const float* gb1 = (const float*)(ws + B1_OFF);
  const unsigned short* xbf = (const unsigned short*)(ws + XBF_OFF);
  float* h0f = (float*)(ws + H0F_OFF);
  float* c0f = (float*)(ws + C0F_OFF);
  float* h1f = (float*)(ws + H1F_OFF);
  float* c1f = (float*)(ws + C1F_OFF);

  const int tid = threadIdx.x;
  const int wave = tid >> 5, lane = tid & 31;
  const int lo = lane & 15, hi = lane >> 4, hi8 = hi << 3, hi16 = hi << 4;
  const int b0row = blockIdx.x * BT;

  // stage enc L0 weights into LDS with row pad 160 -> 168
  for (int c = tid; c < 512 * 20; c += 256) {
    int n = c / 20, j = (c % 20) * 8;
    *(Q16*)(sW0 + n * 168 + j) = *(const Q16*)(gW0 + n * 160 + j);
  }
  for (int i = tid; i < 512; i += 256) { sB0[i] = gb0[i]; sB1[i] = gb1[i]; }
  Q16 zq; zq.x = zq.y = zq.z = zq.w = 0u;
  for (int c = tid; c < 64 * 33; c += 256) *(Q16*)(sCat + c * 8) = zq;   // h0|h1 = 0
  // zero only the pad columns (16..31) of both x buffers; data cols come from async copies
  for (int c = tid; c < 2 * 64 * 2; c += 256) {
    int bufi = c >> 7, r = (c >> 1) & 63, half = c & 1;
    *(Q16*)(sXZ + bufi * (64 * 32) + r * 32 + 16 + half * 8) = zq;
  }
  __syncthreads();

  // prologue: kick off async copy of x_{t=0} into buffer 0
  if (tid < 128) {
    int r = tid >> 1, half = tid & 1;
    async_copy16(xbf + ((size_t)(b0row + r) * TT + 0) * INF + half * 8,
                 sXZ + r * 32 + half * 8);
  }

  int nb[4]; float bl0[4], bl1[4];
#pragma unroll
  for (int g = 0; g < 4; ++g) {
    nb[g] = 16 * wave + 128 * g;
    bl0[g] = sB0[nb[g] + lo];
    bl1[g] = sB1[nb[g] + lo];
  }
  const int hcol = 16 * wave + lo;

  v8f c0[4], c1[4];
#pragma unroll
  for (int mt = 0; mt < 4; ++mt) { c0[mt] = vzero(); c1[mt] = vzero(); }

#pragma unroll 1
  for (int t = 0; t < TT; ++t) {
    const unsigned short* xbuf = sXZ + (t & 1) * (64 * 32);
    wait_async();          // x_t copy (issued last iteration / prologue) complete
    __syncthreads();
    // prefetch x_{t+1} into the other buffer (not read by anyone this step)
    if (t + 1 < TT && tid < 128) {
      int r = tid >> 1, half = tid & 1;
      async_copy16(xbf + ((size_t)(b0row + r) * TT + (t + 1)) * INF + half * 8,
                   sXZ + ((t + 1) & 1) * (64 * 32) + r * 32 + half * 8);
    }

    // ---- layer 0: gates = [h0 | x | 0pad] @ Wc0^T ----
    v8f acc[4][4];
#pragma unroll
    for (int mt = 0; mt < 4; ++mt)
#pragma unroll
      for (int g = 0; g < 4; ++g) acc[mt][g] = vzero();
#pragma unroll
    for (int kt = 0; kt < 5; ++kt) {
      v16bf a[4];
#pragma unroll
      for (int mt = 0; mt < 4; ++mt) {
        int row = mt * 16 + lo;
        if (kt < 4) a[mt] = loadAfrag(sCat, 264, row, kt * 32 + hi8);
        else        a[mt] = loadAfrag(xbuf, 32, row, hi8);
      }
#pragma unroll
      for (int g = 0; g < 4; ++g) {
        v16bf b = loadBfrag(sW0, 168, nb[g] + lo, kt * 32 + hi8);
#pragma unroll
        for (int mt = 0; mt < 4; ++mt) acc[mt][g] = wmma_bf16(a[mt], b, acc[mt][g]);
      }
    }
    // LSTM cell layer 0
    float hv[4][8];
#pragma unroll
    for (int mt = 0; mt < 4; ++mt)
#pragma unroll
      for (int e = 0; e < 8; ++e) {
        float ig = sigm(acc[mt][0][e] + bl0[0]);
        float fg = sigm(acc[mt][1][e] + bl0[1]);
        float gg = tanh_f(acc[mt][2][e] + bl0[2]);
        float og = sigm(acc[mt][3][e] + bl0[3]);
        float cc = fg * c0[mt][e] + ig * gg;
        c0[mt][e] = cc;
        hv[mt][e] = og * tanh_f(cc);
      }
    __syncthreads();
#pragma unroll
    for (int mt = 0; mt < 4; ++mt)
#pragma unroll
      for (int e = 0; e < 8; ++e)
        sCat[(mt * 16 + e + hi8) * 264 + hcol] = f2bf(hv[mt][e]);
    if (t == TT - 1) {
#pragma unroll
      for (int mt = 0; mt < 4; ++mt)
#pragma unroll
        for (int e = 0; e < 8; ++e) {
          size_t grow = (size_t)(b0row + mt * 16 + e + hi8);
          h0f[grow * HH + hcol] = hv[mt][e];
          c0f[grow * HH + hcol] = c0[mt][e];
        }
    }
    __syncthreads();

    // ---- layer 1: gates = [h0 | h1] @ Wc1^T  (weights streamed from L2) ----
#pragma unroll
    for (int mt = 0; mt < 4; ++mt)
#pragma unroll
      for (int g = 0; g < 4; ++g) acc[mt][g] = vzero();
#pragma unroll
    for (int kt = 0; kt < 8; ++kt) {
      v16bf a[4];
#pragma unroll
      for (int mt = 0; mt < 4; ++mt) a[mt] = loadAfrag(sCat, 264, mt * 16 + lo, kt * 32 + hi8);
#pragma unroll
      for (int g = 0; g < 4; ++g) {
        v16bf b = loadBfrag(gW1, 256, nb[g] + lo, kt * 32 + hi8);
#pragma unroll
        for (int mt = 0; mt < 4; ++mt) acc[mt][g] = wmma_bf16(a[mt], b, acc[mt][g]);
      }
    }
#pragma unroll
    for (int mt = 0; mt < 4; ++mt)
#pragma unroll
      for (int e = 0; e < 8; ++e) {
        float ig = sigm(acc[mt][0][e] + bl1[0]);
        float fg = sigm(acc[mt][1][e] + bl1[1]);
        float gg = tanh_f(acc[mt][2][e] + bl1[2]);
        float og = sigm(acc[mt][3][e] + bl1[3]);
        float cc = fg * c1[mt][e] + ig * gg;
        c1[mt][e] = cc;
        hv[mt][e] = og * tanh_f(cc);
      }
    if (t == TT - 1) {
#pragma unroll
      for (int mt = 0; mt < 4; ++mt)
#pragma unroll
        for (int e = 0; e < 8; ++e) {
          size_t grow = (size_t)(b0row + mt * 16 + e + hi8);
          h1f[grow * HH + hcol] = hv[mt][e];
          c1f[grow * HH + hcol] = c1[mt][e];
        }
    }
    __syncthreads();
#pragma unroll
    for (int mt = 0; mt < 4; ++mt)
#pragma unroll
      for (int e = 0; e < 8; ++e)
        sCat[(mt * 16 + e + hi8) * 264 + 128 + hcol] = f2bf(hv[mt][e]);
  }
}

// ---------------- fused 2-layer LSTM decoder + FC ----------------
__global__ __launch_bounds__(256) void dec_kernel(const float* __restrict__ x,
                                                  const float* __restrict__ dWih0,
                                                  const float* __restrict__ fcW,
                                                  const float* __restrict__ fcb,
                                                  char* __restrict__ ws,
                                                  float* __restrict__ out) {
  extern __shared__ char smem[];
  unsigned short* sW1  = (unsigned short*)(smem + DEC_SW1);
  unsigned short* sCat = (unsigned short*)(smem + DEC_SCAT);
  float* sOut = (float*)(smem + DEC_SOUT);
  float* sFc  = (float*)(smem + DEC_SFC);
  float* sFcb = (float*)(smem + DEC_SFCB);
  float* sB0  = (float*)(smem + DEC_SB0);
  float* sB1  = (float*)(smem + DEC_SB1);

  const unsigned short* gDh0 = (const unsigned short*)(ws + DHH0_OFF);
  const unsigned short* gDc1 = (const unsigned short*)(ws + DC1_OFF);
  const float* gdb0 = (const float*)(ws + DB0_OFF);
  const float* gdb1 = (const float*)(ws + DB1_OFF);
  const float* h0f = (const float*)(ws + H0F_OFF);
  const float* c0f = (const float*)(ws + C0F_OFF);
  const float* h1f = (const float*)(ws + H1F_OFF);
  const float* c1f = (const float*)(ws + C1F_OFF);

  const int tid = threadIdx.x;
  const int wave = tid >> 5, lane = tid & 31;
  const int lo = lane & 15, hi = lane >> 4, hi8 = hi << 3, hi16 = hi << 4;
  const int b0row = blockIdx.x * BT;

  // stage dec L1 concat weights 256 -> 264 stride
  for (int c = tid; c < 512 * 32; c += 256) {
    int n = c >> 5, j = (c & 31) * 8;
    *(Q16*)(sW1 + n * 264 + j) = *(const Q16*)(gDc1 + n * 256 + j);
  }
  for (int i = tid; i < 512; i += 256) { sB0[i] = gdb0[i]; sB1[i] = gdb1[i]; }
  for (int i = tid; i < 256; i += 256) sFc[i] = fcW[i];
  if (tid < 2) sFcb[tid] = fcb[tid];
  // init h0|h1 tiles from encoder finals
  for (int i = tid; i < 64 * 128; i += 256) {
    int r = i >> 7, k = i & 127;
    sCat[r * 264 + k]       = f2bf(h0f[(size_t)(b0row + r) * HH + k]);
    sCat[r * 264 + 128 + k] = f2bf(h1f[(size_t)(b0row + r) * HH + k]);
  }
  // dec_in = x[:, -1, :2]
  if (tid < 128) {
    int r = tid >> 1, j = tid & 1;
    sOut[r * 2 + j] = x[((size_t)(b0row + r) * TT + (TT - 1)) * INF + j];
  }
  __syncthreads();

  int nb[4]; float bl0[4], bl1[4], wi0[4], wi1[4];
#pragma unroll
  for (int g = 0; g < 4; ++g) {
    nb[g] = 16 * wave + 128 * g;
    bl0[g] = sB0[nb[g] + lo];
    bl1[g] = sB1[nb[g] + lo];
    wi0[g] = dWih0[(nb[g] + lo) * 2 + 0];
    wi1[g] = dWih0[(nb[g] + lo) * 2 + 1];
  }
  const int hcol = 16 * wave + lo;

  v8f c0[4], c1[4];
#pragma unroll
  for (int mt = 0; mt < 4; ++mt)
#pragma unroll
    for (int e = 0; e < 8; ++e) {
      size_t grow = (size_t)(b0row + mt * 16 + e + hi8);
      c0[mt][e] = c0f[grow * HH + hcol];
      c1[mt][e] = c1f[grow * HH + hcol];
    }

#pragma unroll 1
  for (int s = 0; s < TGTN; ++s) {
    __syncthreads();
    // ---- layer 0: gates = inp(2, scalar FMA) + dh0 @ dWhh0^T (weights from L2) ----
    v8f acc[4][4];
#pragma unroll
    for (int mt = 0; mt < 4; ++mt)
#pragma unroll
      for (int g = 0; g < 4; ++g) acc[mt][g] = vzero();
#pragma unroll
    for (int kt = 0; kt < 4; ++kt) {
      v16bf a[4];
#pragma unroll
      for (int mt = 0; mt < 4; ++mt) a[mt] = loadAfrag(sCat, 264, mt * 16 + lo, kt * 32 + hi8);
#pragma unroll
      for (int g = 0; g < 4; ++g) {
        v16bf b = loadBfrag(gDh0, 128, nb[g] + lo, kt * 32 + hi8);
#pragma unroll
        for (int mt = 0; mt < 4; ++mt) acc[mt][g] = wmma_bf16(a[mt], b, acc[mt][g]);
      }
    }
    float hv[4][8];
#pragma unroll
    for (int mt = 0; mt < 4; ++mt)
#pragma unroll
      for (int e = 0; e < 8; ++e) {
        int rrow = mt * 16 + e + hi8;
        float i0 = sOut[rrow * 2 + 0], i1 = sOut[rrow * 2 + 1];
        float ig = sigm(acc[mt][0][e] + bl0[0] + wi0[0] * i0 + wi1[0] * i1);
        float fg = sigm(acc[mt][1][e] + bl0[1] + wi0[1] * i0 + wi1[1] * i1);
        float gg = tanh_f(acc[mt][2][e] + bl0[2] + wi0[2] * i0 + wi1[2] * i1);
        float og = sigm(acc[mt][3][e] + bl0[3] + wi0[3] * i0 + wi1[3] * i1);
        float cc = fg * c0[mt][e] + ig * gg;
        c0[mt][e] = cc;
        hv[mt][e] = og * tanh_f(cc);
      }
    __syncthreads();
#pragma unroll
    for (int mt = 0; mt < 4; ++mt)
#pragma unroll
      for (int e = 0; e < 8; ++e)
        sCat[(mt * 16 + e + hi8) * 264 + hcol] = f2bf(hv[mt][e]);
    __syncthreads();

    // ---- layer 1: gates = [dh0 | dh1] @ Dc1^T (weights in LDS) ----
#pragma unroll
    for (int mt = 0; mt < 4; ++mt)
#pragma unroll
      for (int g = 0; g < 4; ++g) acc[mt][g] = vzero();
#pragma unroll
    for (int kt = 0; kt < 8; ++kt) {
      v16bf a[4];
#pragma unroll
      for (int mt = 0; mt < 4; ++mt) a[mt] = loadAfrag(sCat, 264, mt * 16 + lo, kt * 32 + hi8);
#pragma unroll
      for (int g = 0; g < 4; ++g) {
        v16bf b = loadBfrag(sW1, 264, nb[g] + lo, kt * 32 + hi8);
#pragma unroll
        for (int mt = 0; mt < 4; ++mt) acc[mt][g] = wmma_bf16(a[mt], b, acc[mt][g]);
      }
    }
#pragma unroll
    for (int mt = 0; mt < 4; ++mt)
#pragma unroll
      for (int e = 0; e < 8; ++e) {
        float ig = sigm(acc[mt][0][e] + bl1[0]);
        float fg = sigm(acc[mt][1][e] + bl1[1]);
        float gg = tanh_f(acc[mt][2][e] + bl1[2]);
        float og = sigm(acc[mt][3][e] + bl1[3]);
        float cc = fg * c1[mt][e] + ig * gg;
        c1[mt][e] = cc;
        hv[mt][e] = og * tanh_f(cc);
      }
    __syncthreads();
#pragma unroll
    for (int mt = 0; mt < 4; ++mt)
#pragma unroll
      for (int e = 0; e < 8; ++e)
        sCat[(mt * 16 + e + hi8) * 264 + 128 + hcol] = f2bf(hv[mt][e]);
    __syncthreads();

    // ---- FC: out = dh1 @ fcW^T + fcb; becomes next input ----
    if (tid < 128) {
      int r = tid >> 1, j = tid & 1;
      const unsigned short* hp = sCat + r * 264 + 128;
      const float* wp = sFc + j * 128;
      float accf = sFcb[j];
#pragma unroll 8
      for (int k = 0; k < 128; ++k) accf += bf2f(hp[k]) * wp[k];
      sOut[r * 2 + j] = accf;
      out[((size_t)(b0row + r) * TGTN + s) * OUTD + j] = accf;
    }
  }
}

extern "C" void kernel_launch(void* const* d_in, const int* in_sizes, int n_in,
                              void* d_out, int out_size, void* d_ws, size_t ws_size,
                              hipStream_t stream) {
  const float* x     = (const float*)d_in[0];
  const float* eWih0 = (const float*)d_in[2];
  const float* eWhh0 = (const float*)d_in[3];
  const float* ebih0 = (const float*)d_in[4];
  const float* ebhh0 = (const float*)d_in[5];
  const float* eWih1 = (const float*)d_in[6];
  const float* eWhh1 = (const float*)d_in[7];
  const float* ebih1 = (const float*)d_in[8];
  const float* ebhh1 = (const float*)d_in[9];
  const float* dWih0 = (const float*)d_in[10];
  const float* dWhh0 = (const float*)d_in[11];
  const float* dbih0 = (const float*)d_in[12];
  const float* dbhh0 = (const float*)d_in[13];
  const float* dWih1 = (const float*)d_in[14];
  const float* dWhh1 = (const float*)d_in[15];
  const float* dbih1 = (const float*)d_in[16];
  const float* dbhh1 = (const float*)d_in[17];
  const float* fcW   = (const float*)d_in[18];
  const float* fcb   = (const float*)d_in[19];
  char* ws = (char*)d_ws;
  float* out = (float*)d_out;

  (void)in_sizes; (void)n_in; (void)out_size; (void)ws_size;

  (void)hipFuncSetAttribute((const void*)enc_kernel, hipFuncAttributeMaxDynamicSharedMemorySize, ENC_LDS);
  (void)hipFuncSetAttribute((const void*)dec_kernel, hipFuncAttributeMaxDynamicSharedMemorySize, DEC_LDS);

  prep_kernel<<<dim3(256), 256, 0, stream>>>(x,
                                             eWih0, eWhh0, ebih0, ebhh0,
                                             eWih1, eWhh1, ebih1, ebhh1,
                                             dWhh0, dWih1, dWhh1,
                                             dbih0, dbhh0, dbih1, dbhh1, ws);
  enc_kernel<<<dim3(BB / BT), 256, ENC_LDS, stream>>>(ws);
  dec_kernel<<<dim3(BB / BT), 256, DEC_LDS, stream>>>(x, dWih0, fcW, fcb, ws, out);
}
